// TinyMPNCOVResNetBackbone_12360915878294
// MI455X (gfx1250) — matrix operand order repaired and checked by reading, not compiled, not run on hardware
//
#include <hip/hip_runtime.h>
#include <hip/hip_bf16.h>

// ---------------------------------------------------------------------------
// Types for CDNA5 WMMA fragments (wave32)
// ---------------------------------------------------------------------------
typedef __attribute__((ext_vector_type(16))) __bf16 v16bf;
typedef __attribute__((ext_vector_type(8)))  float  v8f;
typedef __attribute__((ext_vector_type(4)))  float  v4f;
typedef __attribute__((ext_vector_type(2)))  float  v2f;

static __device__ __forceinline__ unsigned short f32_to_bf16_bits(float f) {
    union { float f; unsigned u; } in; in.f = f;
    unsigned r = in.u + 0x7FFFu + ((in.u >> 16) & 1u);   // round-to-nearest-even
    return (unsigned short)(r >> 16);
}

// ---------------------------------------------------------------------------
// Weight pre-pack: bake the CDNA5 A-fragment (16x32 bf16) lane/element
// interleave into [ctile][kchunk][lane][16] bf16 so the conv hot loop loads
// A as one contiguous v16bf per lane. K reordered as k = (kh*KW+kw)*Cin + ci.
// ---------------------------------------------------------------------------
__global__ void pack_weights_kernel(const float* __restrict__ w, unsigned short* __restrict__ wpk,
                                    int Cout, int Cin, int log2Cin, int KK, int kchunks)
{
    int idx = blockIdx.x * blockDim.x + threadIdx.x;
    int total = (Cout / 16) * kchunks * 512;
    if (idx >= total) return;
    int e    = idx & 15;
    int lane = (idx >> 4) & 31;
    int g    = idx >> 9;
    int kc   = g % kchunks;
    int ct   = g / kchunks;
    int half = lane >> 4, l15 = lane & 15;
    int v = e >> 1, bit = e & 1;
    int kl = 16 * half + 2 * v + bit;  // A-frag K contiguous: k = 16*half + e
    int k  = kc * 32 + kl;
    int m  = ct * 16 + l15;
    int Ktot = Cin * KK;
    float val = 0.f;
    if (k < Ktot) {
        int ci = k & (Cin - 1);
        int sp = k >> log2Cin;                // = kh*KW+kw
        val = w[((size_t)m * Cin + ci) * KK + sp];
    }
    wpk[idx] = f32_to_bf16_bits(val);
}

// ---------------------------------------------------------------------------
// LDS-staged fused conv + BN (+residual)(+ReLU) implicit GEMM.
// 128 threads = 4 waves, one image, 64 output pixels (WROWS full rows).
// Input patch staged to LDS (bf16, channel-innermost, zeroed halo).
// Branchless tails: out-of-range K selects a zeroed, 32B-aligned LDS slot.
// Epilogue: vector (v8f) BN loads, hoisted residual branch, base+rI*P stores.
// ---------------------------------------------------------------------------
template<int KW, bool SMALL>
__global__ __launch_bounds__(128)
void conv_bn_wmma(const float* __restrict__ in, const v16bf* __restrict__ wpk,
                  const float* __restrict__ scale, const float* __restrict__ shift,
                  const float* __restrict__ residual, float* __restrict__ out,
                  int nblk, int Cin, int log2Cin, int Hin, int Win, int log2Win,
                  int Cout, int stride, int pad,
                  int Wout, int log2Wout, int P, int WROWS, int rows_in,
                  int relu, int kchunks, int Ktot, int zslot)
{
    extern __shared__ unsigned short tile[];   // [rows_in][Win+2][Cin] bf16 + zero slot

    const int b    = blockIdx.x / nblk;
    const int blk  = blockIdx.x % nblk;
    const int row0 = blk * WROWS;
    const int pix0 = blk * 64;
    const int tid  = threadIdx.x;
    const int pitchC = Win + 2;

    // ---- zero LDS tile + zero slot ----
    const int totalL = zslot + 16;
    for (int i = tid; i < totalL; i += 128) tile[i] = 0;
    __syncthreads();

    // ---- stage valid input (coalesced along iw; pure shift/mask indexing) ----
    const int totalV = Cin * rows_in * Win;
    const int ih0 = row0 * stride - pad;
    for (int i = tid; i < totalV; i += 128) {
        int iw = i & (Win - 1);
        int t2 = i >> log2Win;
        int ci = t2 & (Cin - 1);
        int rs = t2 >> log2Cin;
        int ih = ih0 + rs;
        if (ih >= 0 && ih < Hin) {
            float v = in[(((size_t)b * Cin + ci) * Hin + ih) * Win + iw];
            tile[(rs * pitchC + (iw + 1)) * Cin + ci] = f32_to_bf16_bits(v);
        }
    }
    __syncthreads();

    // ---- compute ----
    const int wave = tid >> 5;
    const int lane = tid & 31;
    const int half = lane >> 4;
    const int l15  = lane & 15;

    const int p    = pix0 + wave * 16 + l15;       // global pixel (< P by construction)
    const int oh_l = (p >> log2Wout) - row0;
    const int ow   = p & (Wout - 1);
    const int rowb = oh_l * stride;
    const int colb = ow * stride - pad + 1;
    const int ctiles = Cout >> 4;

    for (int ct = 0; ct < ctiles; ++ct) {
        v8f acc = {};
        const v16bf* wa = wpk + (size_t)ct * kchunks * 32 + lane;
        for (int kc = 0; kc < kchunks; ++kc) {
            v16bf afrag = wa[kc * 32];
            __builtin_prefetch(&wa[(kc + 1) * 32], 0, 1);   // speculative; safe past end

            const int k0h = kc * 32 + 16 * half;
            v16bf bfrag;
            if (!SMALL) {
                // one spatial tap, 16 contiguous channels -> vector LDS read
                int sp  = k0h >> log2Cin;
                int ci0 = k0h & (Cin - 1);
                int rr, cc;
                if (KW == 3) { rr = sp / 3; cc = sp - rr * 3; }
                else         { rr = 0;      cc = 0; }
                int off = ((rowb + rr) * pitchC + colb + cc) * Cin + ci0;
                off = (k0h < Ktot) ? off : zslot;            // branchless tail
                bfrag = *(const v16bf*)(tile + off);
            } else {
                // small-Cin (conv1): per-element, unpredicated LDS reads
#pragma unroll
                for (int e = 0; e < 16; ++e) {
                    int k  = k0h + e;
                    int ci = k & (Cin - 1);
                    int sp = k >> log2Cin;
                    int rr = sp / 3, cc = sp - rr * 3;
                    int off = ((rowb + rr) * pitchC + colb + cc) * Cin + ci;
                    off = (k < Ktot) ? off : zslot;          // branchless
                    union { unsigned short s; __bf16 b; } u;
                    u.s = tile[off];
                    bfrag[e] = u.b;
                }
            }
            acc = __builtin_amdgcn_wmma_f32_16x16x32_bf16(false, afrag, false, bfrag,
                                                          (short)0, acc, false, false);
        }
        // ---- epilogue: vector BN loads, hoisted residual branch ----
        const int mbase = ct * 16 + 8 * half;             // 8 consecutive channels
        v8f sc8 = *(const v8f*)(scale + mbase);           // 2x global_load_b128
        v8f sh8 = *(const v8f*)(shift + mbase);
        float* outp = out + ((size_t)b * Cout + mbase) * (size_t)P + p;
        if (residual) {
            const float* resp = residual + ((size_t)b * Cout + mbase) * (size_t)P + p;
#pragma unroll
            for (int rI = 0; rI < 8; ++rI) {
                float v = acc[rI] * sc8[rI] + sh8[rI] + resp[(size_t)rI * P];
                if (relu) v = fmaxf(v, 0.f);
                outp[(size_t)rI * P] = v;
            }
        } else {
#pragma unroll
            for (int rI = 0; rI < 8; ++rI) {
                float v = acc[rI] * sc8[rI] + sh8[rI];
                if (relu) v = fmaxf(v, 0.f);
                outp[(size_t)rI * P] = v;
            }
        }
    }
}

// ---------------------------------------------------------------------------
// Batched fp32 GEMM on v_wmma_f32_16x16x4_f32 (covpool / Newton-Schulz).
// M,N,K multiples of 16 -> no bounds checks. K-loop unrolled 4x.
// BVEC=1: B k-contiguous (covpool Xc^T); BVEC=0: B k-strided (row-major NS).
// mode 0: D = alpha*(A@B)*(bscale?bscale[b]:1);  mode 1: D = 0.5*(3I - A@B)
// ---------------------------------------------------------------------------
template<int BVEC>
__global__ __launch_bounds__(32)
void gemm_ns_wmma(const float* __restrict__ A, const float* __restrict__ Bm,
                  float* __restrict__ D,
                  int N, int K,
                  long long sAb, long long sBb, long long sDb,
                  int lda, int Bs_k, int Bs_n,
                  int mode, float alpha, const float* __restrict__ bscale)
{
    const int b      = blockIdx.y;
    const int ntiles = N >> 4;
    const int mt     = blockIdx.x / ntiles;
    const int nt     = blockIdx.x % ntiles;
    const int lane   = threadIdx.x & 31;
    const int half   = lane >> 4;
    const int l15    = lane & 15;

    const float* Ab = A  + (size_t)b * sAb + (size_t)(mt * 16 + l15) * lda;
    const float* Bb = Bm + (size_t)b * sBb + (size_t)(nt * 16 + l15) * Bs_n;
    const int n = nt * 16 + l15;

    v8f acc = {};
#pragma unroll 4
    for (int k0 = 0; k0 < K; k0 += 4) {
        int k = k0 + 2 * half;
        v2f af = *(const v2f*)(Ab + k);
        v2f bf;
        if (BVEC) {
            bf = *(const v2f*)(Bb + k);
        } else {
            bf[0] = Bb[(size_t)k * Bs_k];
            bf[1] = Bb[(size_t)(k + 1) * Bs_k];
        }
        acc = __builtin_amdgcn_wmma_f32_16x16x4_f32(false, af, false, bf,
                                                    (short)0, acc, false, false);
    }

    const int mbase = mt * 16 + 8 * half;
    float* Dp = D + (size_t)b * sDb + (size_t)mbase * N + n;
    if (mode == 1) {
#pragma unroll
        for (int rI = 0; rI < 8; ++rI) {
            int mm = mbase + rI;
            Dp[(size_t)rI * N] = 0.5f * ((mm == n ? 3.0f : 0.0f) - acc[rI]);
        }
    } else {
        float s = alpha;
        if (bscale) s *= bscale[b];
#pragma unroll
        for (int rI = 0; rI < 8; ++rI)
            Dp[(size_t)rI * N] = acc[rI] * s;
    }
}

// ---------------------------------------------------------------------------
// Small helper kernels
// ---------------------------------------------------------------------------
__global__ void fold_bn_kernel(const float* __restrict__ g, const float* __restrict__ be,
                               const float* __restrict__ m, const float* __restrict__ v,
                               float* __restrict__ scale, float* __restrict__ shift, int n)
{
    int i = blockIdx.x * blockDim.x + threadIdx.x;
    if (i < n) {
        float inv = g[i] * rsqrtf(v[i] + 1e-5f);
        scale[i] = inv;
        shift[i] = be[i] - m[i] * inv;
    }
}

__global__ void center_kernel(const float* __restrict__ in, float* __restrict__ out, int C, int P)
{
    int b = blockIdx.y, c = blockIdx.x, t = threadIdx.x;
    __shared__ float red[64];
    const float* p = in + ((size_t)b * C + c) * (size_t)P;
    float v = p[t];
    red[t] = v;
    __syncthreads();
    for (int s = blockDim.x >> 1; s > 0; s >>= 1) {
        if (t < s) red[t] += red[t + s];
        __syncthreads();
    }
    float mean = red[0] / (float)P;
    out[((size_t)b * C + c) * (size_t)P + t] = v - mean;
}

__global__ void trace_kernel(const float* __restrict__ A, int C,
                             float* __restrict__ invN, float* __restrict__ sqrtN)
{
    int b = blockIdx.x, t = threadIdx.x;
    __shared__ float red[128];
    float v = (t < C) ? A[((size_t)b * C + t) * (size_t)C + t] : 0.f;
    red[t] = v;
    __syncthreads();
    for (int s = blockDim.x >> 1; s > 0; s >>= 1) {
        if (t < s) red[t] += red[t + s];
        __syncthreads();
    }
    if (t == 0) { float tr = red[0]; invN[b] = 1.0f / tr; sqrtN[b] = sqrtf(tr); }
}

__global__ void ns_prep_kernel(const float* __restrict__ A, const float* __restrict__ invN,
                               float* __restrict__ An, float* __restrict__ Z,
                               int C, long long total)
{
    long long i = (long long)blockIdx.x * blockDim.x + threadIdx.x;
    if (i >= total) return;
    long long cc = (long long)C * C;
    int b   = (int)(i / cc);
    long long r = i % cc;
    int row = (int)(r / C), col = (int)(r % C);
    float an = A[i] * invN[b];
    An[i] = an;
    Z[i]  = 0.5f * ((row == col ? 3.f : 0.f) - an);
}

__global__ void triuvec_kernel(const float* __restrict__ A, float* __restrict__ out,
                               int C, int T, int Bn)
{
    int idx = blockIdx.x * blockDim.x + threadIdx.x;
    if (idx >= Bn * T) return;
    int b = idx / T, t = idx % T;
    float twoC1 = 2.f * C + 1.f;
    int i = (int)((twoC1 - sqrtf(twoC1 * twoC1 - 8.f * (float)t)) * 0.5f);
    if (i < 0) i = 0;
    if (i > C - 1) i = C - 1;
    while (i > 0 && (long long)i * C - (long long)i * (i - 1) / 2 > t) --i;
    while ((long long)(i + 1) * C - (long long)(i + 1) * i / 2 <= t) ++i;
    int base = i * C - i * (i - 1) / 2;
    int j = i + (t - base);
    out[(size_t)b * T + t] = A[((size_t)b * C + i) * (size_t)C + j];
}

__global__ void fc_sigmoid_kernel(const float* __restrict__ v, const float* __restrict__ W,
                                  const float* __restrict__ bias, float* __restrict__ s,
                                  int B, int O, int Kd)
{
    int idx = blockIdx.x * blockDim.x + threadIdx.x;
    if (idx >= B * O) return;
    int b = idx / O, o = idx % O;
    const float* vp = v + (size_t)b * Kd;
    const float* wp = W + (size_t)o * Kd;
    float acc = bias[o];
    for (int k = 0; k < Kd; k += 4) {                    // Kd % 4 == 0 (2080)
        v4f a = *(const v4f*)(vp + k);
        v4f w4 = *(const v4f*)(wp + k);
        acc += a[0] * w4[0] + a[1] * w4[1] + a[2] * w4[2] + a[3] * w4[3];
    }
    s[idx] = 1.f / (1.f + __expf(-acc));
}

__global__ void gate_kernel(const float* __restrict__ x, const float* __restrict__ s,
                            float* __restrict__ out, int C, int P, long long total)
{
    long long i = (long long)blockIdx.x * blockDim.x + threadIdx.x;
    if (i >= total) return;
    int b = (int)(i / ((long long)C * P));
    int c = (int)((i / P) % C);
    out[i] = x[i] * s[(size_t)b * C + c];
}

// ---------------------------------------------------------------------------
// Host orchestration
// ---------------------------------------------------------------------------
static inline int ilog2(int x) { int l = 0; while ((1 << l) < x) ++l; return l; }

extern "C" void kernel_launch(void* const* d_in, const int* in_sizes, int n_in,
                              void* d_out, int out_size, void* d_ws, size_t ws_size,
                              hipStream_t stream)
{
    (void)in_sizes; (void)n_in; (void)out_size; (void)ws_size;
    const int B = 128;

    // --- input index map (jax pytree: dict keys sorted alphabetically) ---
    const int IDX_ATT_BN = 0, IDX_ATT_FCB = 4, IDX_ATT_FCW = 5, IDX_ATT_W = 6;
    const int IDX_BN1 = 7, IDX_CONV1 = 11, IDX_REDUCE_BN = 107, IDX_REDUCE_W = 111, IDX_X = 112;

    float* W = (float*)d_ws;
    size_t off = 0;
    auto alloc = [&](size_t n) { float* p = W + off; off += (n + 63) & ~(size_t)63; return p; };

    const size_t BIG = (size_t)B * 16 * 64 * 64;
    float* p0 = alloc(BIG);
    float* p1 = alloc(BIG);
    float* p2 = alloc(BIG);

    auto foldbn = [&](int gi, int n, float** sc, float** sh) {
        *sc = alloc(n); *sh = alloc(n);
        fold_bn_kernel<<<dim3((n + 63) / 64), dim3(64), 0, stream>>>(
            (const float*)d_in[gi], (const float*)d_in[gi + 1],
            (const float*)d_in[gi + 2], (const float*)d_in[gi + 3], *sc, *sh, n);
    };

    auto conv = [&](const float* in, const float* w, int Cin, int Hin, int Win,
                    int Cout, int KH, int stride, const float* sc, const float* sh,
                    const float* res, int relu, float* out, int* Hout, int* Wout) {
        int Ho = (Hin + stride - 1) / stride, Wo = (Win + stride - 1) / stride;
        int pad = (stride == 1) ? KH / 2 : 0;      // jax SAME (asymmetric for s=2)
        int P = Ho * Wo;
        int KK = KH * KH;
        int Ktot = Cin * KK;
        int kchunks = (Ktot + 31) / 32;
        int ctiles = Cout / 16;
        int lCin = ilog2(Cin), lWo = ilog2(Wo), lWin = ilog2(Win);

        int packElems = ctiles * kchunks * 512;
        unsigned short* wpk = (unsigned short*)alloc((size_t)(packElems + 1) / 2);
        pack_weights_kernel<<<dim3((packElems + 255) / 256), dim3(256), 0, stream>>>(
            w, wpk, Cout, Cin, lCin, KK, kchunks);

        int WROWS = 64 / Wo;
        int rows_in = (WROWS - 1) * stride + KH;
        int pitchC = Win + 2;
        int nblk = P / 64;
        int zslot = (rows_in * pitchC * Cin + 15) & ~15;   // 32B-aligned zero slot
        size_t smem = (size_t)(zslot + 16) * sizeof(unsigned short);
        dim3 grid(B * nblk), blockD(128);

        if (Cin < 16) {
            conv_bn_wmma<3, true><<<grid, blockD, smem, stream>>>(
                in, (const v16bf*)wpk, sc, sh, res, out,
                nblk, Cin, lCin, Hin, Win, lWin, Cout, stride, pad,
                Wo, lWo, P, WROWS, rows_in, relu, kchunks, Ktot, zslot);
        } else if (KH == 3) {
            conv_bn_wmma<3, false><<<grid, blockD, smem, stream>>>(
                in, (const v16bf*)wpk, sc, sh, res, out,
                nblk, Cin, lCin, Hin, Win, lWin, Cout, stride, pad,
                Wo, lWo, P, WROWS, rows_in, relu, kchunks, Ktot, zslot);
        } else {
            conv_bn_wmma<1, false><<<grid, blockD, smem, stream>>>(
                in, (const v16bf*)wpk, sc, sh, res, out,
                nblk, Cin, lCin, Hin, Win, lWin, Cout, stride, pad,
                Wo, lWo, P, WROWS, rows_in, relu, kchunks, Ktot, zslot);
        }
        *Hout = Ho; *Wout = Wo;
    };

    auto gemmCC = [&](const float* A, const float* Bm, float* D, int C,
                      int mode, float alpha, const float* bscale) {
        int tiles = C / 16;
        gemm_ns_wmma<0><<<dim3(tiles * tiles, B), dim3(32), 0, stream>>>(
            A, Bm, D, C, C,
            (long long)C * C, (long long)C * C, (long long)C * C,
            C, C, 1, mode, alpha, bscale);
    };

    // ----- conv1 + bn1 + relu -----
    float *sc, *sh;
    int H = 64, Wd = 64, Ho, Wo;
    foldbn(IDX_BN1, 16, &sc, &sh);
    conv((const float*)d_in[IDX_X], (const float*)d_in[IDX_CONV1],
         4, H, Wd, 16, 3, 1, sc, sh, nullptr, 1, p0, &Ho, &Wo);
    H = Ho; Wd = Wo;

    // ----- residual layers -----
    struct BlkCfg { int cin, cout, down, stride; };
    BlkCfg cfgs[8] = {
        {16, 16, 0, 1}, {16, 16, 0, 1},
        {16, 32, 1, 2}, {32, 32, 0, 1},
        {32, 64, 1, 2}, {64, 64, 0, 1},
        {64, 128, 1, 2}, {128, 128, 0, 1},
    };
    int gi = 12;
    for (int bi = 0; bi < 8; ++bi) {
        BlkCfg c = cfgs[bi];
        int i_bn1 = gi, i_bn2 = gi + 4;
        int i_dbn = -1, i_dw = -1, i_w1, i_w2;
        if (c.down) { i_dbn = gi + 8; i_dw = gi + 12; i_w1 = gi + 13; i_w2 = gi + 14; gi += 15; }
        else        { i_w1 = gi + 8; i_w2 = gi + 9; gi += 10; }

        float *s1, *h1, *s2, *h2;
        foldbn(i_bn1, c.cout, &s1, &h1);
        foldbn(i_bn2, c.cout, &s2, &h2);

        conv(p0, (const float*)d_in[i_w1], c.cin, H, Wd, c.cout, 3, c.stride,
             s1, h1, nullptr, 1, p1, &Ho, &Wo);

        const float* res;
        float* dst;
        if (c.down) {
            float *sd, *hd;
            foldbn(i_dbn, c.cout, &sd, &hd);
            int dh, dw;
            conv(p0, (const float*)d_in[i_dw], c.cin, H, Wd, c.cout, 1, c.stride,
                 sd, hd, nullptr, 0, p2, &dh, &dw);
            res = p2; dst = p0;
        } else {
            res = p0; dst = p2;
        }
        int h2o, w2o;
        conv(p1, (const float*)d_in[i_w2], c.cout, Ho, Wo, c.cout, 3, 1,
             s2, h2, res, 1, dst, &h2o, &w2o);

        H = h2o; Wd = w2o;
        if (!c.down) { float* t = p0; p0 = p2; p2 = t; }
    }
    // p0 = [B,128,8,8]

    // ----- reduce 1x1 conv + bn + relu -----
    foldbn(IDX_REDUCE_BN, 128, &sc, &sh);
    conv(p0, (const float*)d_in[IDX_REDUCE_W], 128, H, Wd, 128, 1, 1,
         sc, sh, nullptr, 1, p1, &Ho, &Wo);
    float* rbuf = p1;
    const int P = Ho * Wo;   // 64

    // ----- attention branch -----
    const int CA = 64, TA = CA * (CA + 1) / 2;
    size_t aCC = (size_t)B * CA * CA;
    float* attAct = alloc((size_t)B * CA * P);
    float* attXc  = alloc((size_t)B * CA * P);
    float* attCov = alloc(aCC);
    float* aAn = alloc(aCC); float* aZ = alloc(aCC); float* aY = alloc(aCC);
    float* aT  = alloc(aCC); float* aY2 = alloc(aCC); float* aZ2 = alloc(aCC);
    float* attTri = alloc((size_t)B * TA);
    float* sgate  = alloc((size_t)B * 128);
    float* invN   = alloc(B); float* sqrtN = alloc(B);

    foldbn(IDX_ATT_BN, CA, &sc, &sh);
    conv(rbuf, (const float*)d_in[IDX_ATT_W], 128, Ho, Wo, CA, 1, 1,
         sc, sh, nullptr, 1, attAct, &Ho, &Wo);

    center_kernel<<<dim3(CA, B), dim3(P), 0, stream>>>(attAct, attXc, CA, P);
    {
        int tiles = CA / 16;
        gemm_ns_wmma<1><<<dim3(tiles * tiles, B), dim3(32), 0, stream>>>(
            attXc, attXc, attCov, CA, P,
            (long long)CA * P, (long long)CA * P, (long long)CA * CA,
            P, 1, P, 0, 1.0f / (float)P, nullptr);
    }

    auto sqrtmNS = [&](const float* cov, int C, float* An, float* Z, float* Y,
                       float* T, float* Y2, float* Z2, float* result) {
        trace_kernel<<<dim3(B), dim3(C), 0, stream>>>(cov, C, invN, sqrtN);
        long long tot = (long long)B * C * C;
        ns_prep_kernel<<<dim3((unsigned)((tot + 255) / 256)), dim3(256), 0, stream>>>(
            cov, invN, An, Z, C, tot);
        gemmCC(An, Z, Y, C, 0, 1.0f, nullptr);
        for (int it = 0; it < 3; ++it) {
            gemmCC(Z, Y, T, C, 1, 1.0f, nullptr);
            gemmCC(Y, T, Y2, C, 0, 1.0f, nullptr);
            gemmCC(T, Z, Z2, C, 0, 1.0f, nullptr);
            float* t;
            t = Y; Y = Y2; Y2 = t;
            t = Z; Z = Z2; Z2 = t;
        }
        gemmCC(Z, Y, T, C, 1, 1.0f, nullptr);
        gemmCC(Y, T, result, C, 0, 1.0f, sqrtN);
    };

    sqrtmNS(attCov, CA, aAn, aZ, aY, aT, aY2, aZ2, aAn);
    triuvec_kernel<<<dim3((B * TA + 255) / 256), dim3(256), 0, stream>>>(aAn, attTri, CA, TA, B);
    fc_sigmoid_kernel<<<dim3((B * 128 + 255) / 256), dim3(256), 0, stream>>>(
        attTri, (const float*)d_in[IDX_ATT_FCW], (const float*)d_in[IDX_ATT_FCB],
        sgate, B, 128, TA);

    // ----- gate + final covpool + sqrtm + triuvec -----
    const int CF = 128, TF = CF * (CF + 1) / 2;
    size_t fCC = (size_t)B * CF * CF;
    float* gated = alloc((size_t)B * CF * P);
    float* gXc   = alloc((size_t)B * CF * P);
    float* fCov  = alloc(fCC);
    float* fAn = alloc(fCC); float* fZ = alloc(fCC); float* fY = alloc(fCC);
    float* fT  = alloc(fCC); float* fY2 = alloc(fCC); float* fZ2 = alloc(fCC);

    long long gtot = (long long)B * CF * P;
    gate_kernel<<<dim3((unsigned)((gtot + 255) / 256)), dim3(256), 0, stream>>>(
        rbuf, sgate, gated, CF, P, gtot);
    center_kernel<<<dim3(CF, B), dim3(P), 0, stream>>>(gated, gXc, CF, P);
    {
        int tiles = CF / 16;
        gemm_ns_wmma<1><<<dim3(tiles * tiles, B), dim3(32), 0, stream>>>(
            gXc, gXc, fCov, CF, P,
            (long long)CF * P, (long long)CF * P, (long long)CF * CF,
            P, 1, P, 0, 1.0f / (float)P, nullptr);
    }
    sqrtmNS(fCov, CF, fAn, fZ, fY, fT, fY2, fZ2, fAn);
    triuvec_kernel<<<dim3((B * TF + 255) / 256), dim3(256), 0, stream>>>(
        fAn, (float*)d_out, CF, TF, B);
}
